// BlackholeBlock_62328565399661
// MI455X (gfx1250) — matrix operand order repaired and checked by reading, not compiled, not run on hardware
//
#include <hip/hip_runtime.h>

// ---------------------------------------------------------------------------
// Types
// ---------------------------------------------------------------------------
typedef __attribute__((ext_vector_type(16))) _Float16 v16h;
typedef __attribute__((ext_vector_type(8)))  _Float16 v8h;
typedef __attribute__((ext_vector_type(4)))  _Float16 v4h;
typedef __attribute__((ext_vector_type(2)))  _Float16 v2h;
typedef __attribute__((ext_vector_type(8)))  float    v8f;
typedef __attribute__((ext_vector_type(4)))  float    v4f;

#define BB   32
#define HH   56
#define CC   256
#define NH8  8
#define HD32 32
#define KCH  16
#define KSPT 28
#define NTOK 49                 // tokens per window
#define BW   2048               // windows = 32*8*8
#define BN   (BW*NH8)           // 16384 head-batches
#define MTOK (BW*NTOK)          // 100352 window-tokens
#define LSEQ (HH*HH)            // 3136
#define MLPH 1024
#define SCALE_F 0.17677669529663687f   // 32^-0.5

__device__ inline float wsum(float v) {
#pragma unroll
  for (int o = 16; o > 0; o >>= 1) v += __shfl_xor(v, o, 32);
  return v;
}
__device__ inline float wmaxr(float v) {
#pragma unroll
  for (int o = 16; o > 0; o >>= 1) v = fmaxf(v, __shfl_xor(v, o, 32));
  return v;
}
__device__ inline float gelu_exact(float x) {
  return 0.5f * x * (1.0f + erff(x * 0.70710678118654752f));
}

// ---------------------------------------------------------------------------
// Weight pre-swizzle: f32 (K,N) row-major  ->  f16 WMMA B-fragment order.
// dst[(((ntile*(K/32) + kt)*32 + lane)*16 + e] = B[kt*32 + (lane>>4)*16 + e]
//                                                 [ntile*16 + (lane&15)]
// so a wave's B fragment for (ntile, kt) is one contiguous 32B load per lane.
// ---------------------------------------------------------------------------
__global__ __launch_bounds__(256) void swizzleB_kernel(const float* __restrict__ src,
                                                       _Float16* __restrict__ dst,
                                                       int K, int N) {
  int idx = blockIdx.x * 256 + threadIdx.x;
  if (idx >= K * N) return;
  int e    = idx & 15;
  int lane = (idx >> 4) & 31;
  int rest = idx >> 9;
  int ks   = K >> 5;
  int kt   = rest % ks;
  int nt   = rest / ks;
  int n = nt * 16 + (lane & 15);
  int k = kt * 32 + (lane >> 4) * 16 + e;
  dst[idx] = (_Float16)src[k * N + n];
}

// ---------------------------------------------------------------------------
// LayerNorm1 + window partition : x (B,L,C) f32 -> xw (MTOK, 256) f16
// one wave per token
// ---------------------------------------------------------------------------
__global__ __launch_bounds__(256) void ln_win_kernel(const float* __restrict__ x,
                                                     const float* __restrict__ g,
                                                     const float* __restrict__ bta,
                                                     _Float16* __restrict__ xw) {
  int wave = threadIdx.x >> 5, lane = threadIdx.x & 31;
  int tt = blockIdx.x * 8 + wave;           // window-token id
  int w = tt / NTOK, n = tt % NTOK;
  int b = w >> 6, wy = (w >> 3) & 7, wx = w & 7;
  int l = (wy * 7 + n / 7) * HH + (wx * 7 + n % 7);
  const float* px = x + ((long)b * LSEQ + l) * CC + lane * 8;
  v4f a0 = ((const v4f*)px)[0];
  v4f a1 = ((const v4f*)px)[1];
  float s = 0.f, ss = 0.f;
#pragma unroll
  for (int i = 0; i < 8; ++i) {
    float v = (i < 4) ? a0[i] : a1[i - 4];
    s += v; ss += v * v;
  }
  s = wsum(s); ss = wsum(ss);
  float mu = s * (1.f / 256.f);
  float var = ss * (1.f / 256.f) - mu * mu;
  float rstd = rsqrtf(var + 1e-5f);
  _Float16* pw = xw + (long)tt * CC + lane * 8;
#pragma unroll
  for (int i = 0; i < 8; ++i) {
    float v = (i < 4) ? a0[i] : a1[i - 4];
    pw[i] = (_Float16)((v - mu) * rstd * g[lane * 8 + i] + bta[lane * 8 + i]);
  }
}

// LayerNorm2 (flat) : xin f32 (MTOK,256) -> xn2 f16
__global__ __launch_bounds__(256) void ln_flat_kernel(const float* __restrict__ x,
                                                      const float* __restrict__ g,
                                                      const float* __restrict__ bta,
                                                      _Float16* __restrict__ xo) {
  int wave = threadIdx.x >> 5, lane = threadIdx.x & 31;
  int t = blockIdx.x * 8 + wave;
  const float* px = x + (long)t * CC + lane * 8;
  v4f a0 = ((const v4f*)px)[0];
  v4f a1 = ((const v4f*)px)[1];
  float s = 0.f, ss = 0.f;
#pragma unroll
  for (int i = 0; i < 8; ++i) {
    float v = (i < 4) ? a0[i] : a1[i - 4];
    s += v; ss += v * v;
  }
  s = wsum(s); ss = wsum(ss);
  float mu = s * (1.f / 256.f);
  float var = ss * (1.f / 256.f) - mu * mu;
  float rstd = rsqrtf(var + 1e-5f);
  _Float16* pw = xo + (long)t * CC + lane * 8;
#pragma unroll
  for (int i = 0; i < 8; ++i) {
    float v = (i < 4) ? a0[i] : a1[i - 4];
    pw[i] = (_Float16)((v - mu) * rstd * g[lane * 8 + i] + bta[lane * 8 + i]);
  }
}

// ---------------------------------------------------------------------------
// WMMA f16 GEMM: C(M,N) = A(M,K) @ B(K,N), f32 accum, functor epilogue.
// block = 8 waves; each wave owns WN 16x16 accumulators (16 x 16*WN output);
// K unrolled x2 -> 2*WN back-to-back v_wmma per staged A tile.
// A staged in LDS (shared by all waves); B fragments read directly from the
// pre-swizzled global weight buffer (L2-resident) as 32B contiguous loads.
// ---------------------------------------------------------------------------
template <class Epi, int WN>
__global__ __launch_bounds__(256) void gemm_k(const _Float16* __restrict__ A,
                                              const _Float16* __restrict__ Bsw,
                                              int K, Epi epi) {
  __shared__ __align__(16) _Float16 As[16 * 68];
  const int tid = threadIdx.x;
  const int wave = tid >> 5, lane = tid & 31;
  const int row0 = blockIdx.x * 16;
  const int col0 = blockIdx.y * (128 * WN);
  const int m = lane & 15, khalf = lane >> 4;
  const int ksteps = K >> 5;
  const int nt0 = (col0 >> 4) + wave * WN;    // this wave's first n-tile

  v8f acc[WN];
#pragma unroll
  for (int t = 0; t < WN; ++t)
#pragma unroll
    for (int i = 0; i < 8; ++i) acc[t][i] = 0.f;

  for (int k0 = 0; k0 < K; k0 += 64) {
    __syncthreads();
    {   // stage A tile 16x64 (4 halves / thread)
      int idx = tid * 4;
      int r = idx >> 6, c = idx & 63;
      *(v4h*)&As[r * 68 + c] = *(const v4h*)&A[(long)(row0 + r) * K + k0 + c];
    }
    __syncthreads();
#pragma unroll
    for (int s = 0; s < 2; ++s) {
      // A fragment (contiguous K order): lanes 0-15 K{0..15}, 16-31 K{16..31}
      v16h a;
#pragma unroll
      for (int j = 0; j < 8; ++j) {
        int kk = s * 32 + khalf * 16 + 2 * j;
        a[2 * j]     = As[m * 68 + kk];
        a[2 * j + 1] = As[m * 68 + kk + 1];
      }
      const int kt = (k0 >> 5) + s;
#pragma unroll
      for (int t = 0; t < WN; ++t) {
        v16h b = *(const v16h*)&Bsw[(((long)(nt0 + t) * ksteps + kt) * 32 + lane) * 16];
        acc[t] = __builtin_amdgcn_wmma_f32_16x16x32_f16(false, a, false, b, (short)0,
                                                        acc[t], false, false);
      }
    }
  }
  // D layout: VGPR r -> row = 8*(lane>>4)+r, col = lane&15
  const int rowb = row0 + khalf * 8;
#pragma unroll
  for (int t = 0; t < WN; ++t)
    epi.store8(rowb, col0 + wave * (16 * WN) + t * 16 + m, acc[t]);
}

// ---------------------------------------------------------------------------
// Epilogue functors (whole 8-row accumulator at once; divide once, then carry)
// ---------------------------------------------------------------------------
struct EpiQ {      // q -> qh (Bw, NH, N, HD) f16 + bias
  const float* bias; _Float16* qh;
  __device__ void store8(int row0, int col, v8f a) const {
    float bb = bias[col];
    int h = col >> 5, d = col & 31;
    int w = row0 / NTOK, n = row0 % NTOK;
#pragma unroll
    for (int r = 0; r < 8; ++r) {
      qh[(((long)(w * NH8 + h) * NTOK + n) << 5) + d] = (_Float16)(a[r] + bb);
      if (++n == NTOK) { n = 0; ++w; }
    }
  }
};
struct EpiK {      // k -> kh (Bw, NH, N, KCH) f16 + bias
  const float* bias; _Float16* kh;
  __device__ void store8(int row0, int col, v8f a) const {
    float bb = bias[col];
    int h = col >> 4, i = col & 15;
    int w = row0 / NTOK, n = row0 % NTOK;
#pragma unroll
    for (int r = 0; r < 8; ++r) {
      kh[(((long)(w * NH8 + h) * NTOK + n) << 4) + i] = (_Float16)(a[r] + bb);
      if (++n == NTOK) { n = 0; ++w; }
    }
  }
};
struct EpiV {      // v -> (Bw, NH, KSPT, HD) f16 + bias
  const float* bias; _Float16* vb;
  __device__ void store8(int row0, int col, v8f a) const {
    float bb = bias[col];
    int h = col >> 5, d = col & 31;
    int w = row0 / KSPT, j = row0 % KSPT;
#pragma unroll
    for (int r = 0; r < 8; ++r) {
      vb[(((long)(w * NH8 + h) * KSPT + j) << 5) + d] = (_Float16)(a[r] + bb);
      if (++j == KSPT) { j = 0; ++w; }
    }
  }
};
struct EpiProj {   // proj + window-reverse + residual -> out f32
  const float* bias; const float* xin; float* out;
  __device__ void store8(int row0, int col, v8f a) const {
    float bb = bias[col];
    int w = row0 / NTOK, n = row0 % NTOK;
#pragma unroll
    for (int r = 0; r < 8; ++r) {
      int b = w >> 6, wy = (w >> 3) & 7, wx = w & 7;
      int l = (wy * 7 + n / 7) * HH + (wx * 7 + n % 7);
      long idx = ((long)b * LSEQ + l) * CC + col;
      out[idx] = xin[idx] + a[r] + bb;
      if (++n == NTOK) { n = 0; ++w; }
    }
  }
};
struct EpiM1 {     // mlp1 + bias + exact gelu -> h f16
  const float* bias; _Float16* hb;
  __device__ void store8(int row0, int col, v8f a) const {
    float bb = bias[col];
    long idx = (long)row0 * MLPH + col;
#pragma unroll
    for (int r = 0; r < 8; ++r) {
      hb[idx] = (_Float16)gelu_exact(a[r] + bb);
      idx += MLPH;
    }
  }
};
struct EpiM2 {     // mlp2 + bias + residual into out f32
  const float* bias; float* out;
  __device__ void store8(int row0, int col, v8f a) const {
    float bb = bias[col];
    long idx = (long)row0 * CC + col;
#pragma unroll
    for (int r = 0; r < 8; ++r) {
      out[idx] += a[r] + bb;
      idx += CC;
    }
  }
};

// ---------------------------------------------------------------------------
// Per-window top-k + gathers. block = window, wave = head.
// ---------------------------------------------------------------------------
__global__ __launch_bounds__(256) void topk_gather_kernel(
    const _Float16* __restrict__ qh, const float* __restrict__ wch,
    const float* __restrict__ bch, _Float16* __restrict__ qcha1,
    _Float16* __restrict__ qspt, _Float16* __restrict__ vsel,
    int* __restrict__ sidxg) {
  __shared__ __align__(16) _Float16 qs[NH8][NTOK * HD32];
  __shared__ float srow[NH8][NTOK];
  __shared__ float feat[NH8][64];
  __shared__ float score[NH8][32];
  __shared__ int   ranks[NH8][NTOK];
  __shared__ int   cidx[NH8][KCH];
  __shared__ int   sidx[NH8][KSPT];
  const int w = blockIdx.x;
  const int h = threadIdx.x >> 5, lane = threadIdx.x & 31;
  const long base = (long)(w * NH8 + h) * NTOK * HD32;

  for (int i = lane; i < NTOK * HD32 / 8; i += 32)
    ((v8h*)qs[h])[i] = ((const v8h*)(qh + base))[i];
  __syncthreads();

  // channel mean / max (lane = channel) + spatial row means
  float mn = 0.f, mx = -3.0e38f;
  for (int n = 0; n < NTOK; ++n) {
    float v = (float)qs[h][n * HD32 + lane];
    mn += v; mx = fmaxf(mx, v);
    float rs = wsum(v);
    if (lane == 0) srow[h][n] = rs * (1.f / 32.f);
  }
  feat[h][lane] = mn * (1.f / 49.f);
  feat[h][32 + lane] = mx;
  __syncthreads();

  // channel score: softmax(gelu(feat @ wch + bch))  (lane = out channel)
  {
    float acc = bch[lane];
#pragma unroll 8
    for (int f = 0; f < 64; ++f) acc += feat[h][f] * wch[f * 32 + lane];
    float gg = gelu_exact(acc);
    float m = wmaxr(gg);
    float e = __expf(gg - m);
    float s = wsum(e);
    score[h][lane] = e / s;
  }
  __syncthreads();

  // top-16 channels via rank (stable tie-break => sorted indices by ballot)
  {
    float sc = score[h][lane];
    int rank = 0;
    for (int c = 0; c < 32; ++c) {
      float o = score[h][c];
      rank += (o > sc) || (o == sc && c < lane);
    }
    bool sel = rank < KCH;
    unsigned mask = (unsigned)__ballot(sel);
    int pos = __popc(mask & ((1u << lane) - 1u));
    if (sel) cidx[h][pos] = lane;
  }
  __syncthreads();

  // spatial ranks over 49 row-means
  for (int n = lane; n < NTOK; n += 32) {
    float sv = srow[h][n];
    int rank = 0;
    for (int m2 = 0; m2 < NTOK; ++m2) {
      float o = srow[h][m2];
      rank += (o > sv) || (o == sv && m2 < n);
    }
    ranks[h][n] = rank;
  }
  __syncthreads();
  for (int n = lane; n < NTOK; n += 32) {
    if (ranks[h][n] < KSPT) {
      int pos = 0;
      for (int m2 = 0; m2 < n; ++m2) pos += (ranks[h][m2] < KSPT);
      sidx[h][pos] = n;
    }
  }
  __syncthreads();
  if (lane < KSPT) sidxg[(w * NH8 + h) * KSPT + lane] = sidx[h][lane];

  // qcha1 (Bw, N, 128): channel gather
  for (int idx = lane; idx < NTOK * KCH; idx += 32) {
    int n = idx >> 4, i = idx & 15;
    qcha1[((long)(w * NTOK + n)) * 128 + h * KCH + i] = qs[h][n * HD32 + cidx[h][i]];
  }
  // qspt (Bn, 28, 16): row+channel gather, pre-scaled
  for (int idx = lane; idx < KSPT * KCH; idx += 32) {
    int j = idx >> 4, i = idx & 15;
    float v = (float)qs[h][sidx[h][j] * HD32 + cidx[h][i]] * SCALE_F;
    qspt[((long)((w * NH8 + h) * KSPT + j)) * KCH + i] = (_Float16)v;
  }
  // vsel (Bw, 28, 256): row gather
  for (int idx = lane; idx < KSPT * HD32; idx += 32) {
    int j = idx >> 5, d = idx & 31;
    vsel[((long)(w * KSPT + j)) * CC + h * HD32 + d] = qs[h][sidx[h][j] * HD32 + d];
  }
}

// ---------------------------------------------------------------------------
// Attention: per head: attn(49x28) = kh(49x16) . qspt^T + rpb ; gate ; softmax;
// o = gate * attn @ v(28x32).  block = window, wave = head.
// ---------------------------------------------------------------------------
__global__ __launch_bounds__(256) void attn_kernel(
    const _Float16* __restrict__ khbuf, const _Float16* __restrict__ qsptb,
    const _Float16* __restrict__ vbuf, const float* __restrict__ rpb,
    const int* __restrict__ sidxg, _Float16* __restrict__ obuf) {
  __shared__ __align__(16) _Float16 ks[NH8][NTOK * KCH];
  __shared__ float pb[NH8][KSPT];
  const int w = blockIdx.x;
  const int h = threadIdx.x >> 5, lane = threadIdx.x & 31;
  const long hb = (long)(w * NH8 + h);

  for (int i = lane; i < NTOK * KCH / 8; i += 32)
    ((v8h*)ks[h])[i] = ((const v8h*)(khbuf + hb * NTOK * KCH))[i];

  float qr[KCH];
  int ry = 0, rx = 0;
  if (lane < KSPT) {
    const _Float16* qp = qsptb + (hb * KSPT + lane) * KCH;
#pragma unroll
    for (int i = 0; i < KCH; ++i) qr[i] = (float)qp[i];
    int sj = sidxg[hb * KSPT + lane];
    ry = sj / 7; rx = sj % 7;
  } else {
#pragma unroll
    for (int i = 0; i < KCH; ++i) qr[i] = 0.f;
  }
  float vr[KSPT];                 // v column `lane` across the 28 keys
#pragma unroll
  for (int j = 0; j < KSPT; ++j) vr[j] = (float)vbuf[(hb * KSPT + j) * HD32 + lane];
  __syncthreads();

  for (int n = 0; n < NTOK; ++n) {
    float sc;
    {
      float a = 0.f;
#pragma unroll
      for (int c = 0; c < KCH; ++c) a += (float)ks[h][n * KCH + c] * qr[c];
      if (lane < KSPT) {
        int rel = (n / 7 - ry + 6) * 13 + (n % 7 - rx + 6);
        sc = a + rpb[rel * NH8 + h];
      } else {
        sc = -3.0e38f;
      }
    }
    float msum = wsum(lane < KSPT ? sc : 0.f);
    float gate = 1.f / (1.f + __expf(-msum * (1.f / 28.f)));
    float mx = wmaxr(sc);
    float e = (lane < KSPT) ? __expf(sc - mx) : 0.f;
    float se = wsum(e);
    if (lane < KSPT) pb[h][lane] = e / se;
    __syncthreads();
    float o = 0.f;
#pragma unroll 7
    for (int j = 0; j < KSPT; ++j) o += pb[h][j] * vr[j];
    o *= gate;
    obuf[((long)(w * NTOK + n)) * CC + h * HD32 + lane] = (_Float16)o;
    __syncthreads();
  }
}

// ---------------------------------------------------------------------------
// Host launcher
// ---------------------------------------------------------------------------
extern "C" void kernel_launch(void* const* d_in, const int* in_sizes, int n_in,
                              void* d_out, int out_size, void* d_ws, size_t ws_size,
                              hipStream_t stream) {
  (void)in_sizes; (void)n_in; (void)out_size; (void)ws_size;
  const float* x   = (const float*)d_in[0];
  const float* n1g = (const float*)d_in[1];
  const float* n1b = (const float*)d_in[2];
  const float* wq  = (const float*)d_in[3];
  const float* bq  = (const float*)d_in[4];
  const float* wk  = (const float*)d_in[5];
  const float* bk  = (const float*)d_in[6];
  const float* wv  = (const float*)d_in[7];
  const float* bv  = (const float*)d_in[8];
  const float* wpj = (const float*)d_in[9];
  const float* bpj = (const float*)d_in[10];
  const float* wch = (const float*)d_in[11];
  const float* bch = (const float*)d_in[12];
  const float* rpb = (const float*)d_in[13];
  const float* n2g = (const float*)d_in[14];
  const float* n2b = (const float*)d_in[15];
  const float* w1  = (const float*)d_in[16];
  const float* b1  = (const float*)d_in[17];
  const float* w2  = (const float*)d_in[18];
  const float* b2  = (const float*)d_in[19];
  float* out = (float*)d_out;

  char* ws = (char*)d_ws;
  size_t off = 0;
  auto alloc = [&](size_t bytes) -> void* {
    off = (off + 255) & ~(size_t)255;
    void* p = ws + off;
    off += bytes;
    return p;
  };
  _Float16* wq_sw  = (_Float16*)alloc((size_t)256 * 256 * 2);
  _Float16* wk_sw  = (_Float16*)alloc((size_t)128 * 128 * 2);
  _Float16* wv_sw  = (_Float16*)alloc((size_t)256 * 256 * 2);
  _Float16* wpj_sw = (_Float16*)alloc((size_t)256 * 256 * 2);
  _Float16* w1_sw  = (_Float16*)alloc((size_t)256 * 1024 * 2);
  _Float16* w2_sw  = (_Float16*)alloc((size_t)1024 * 256 * 2);
  _Float16* xw     = (_Float16*)alloc((size_t)MTOK * CC * 2);
  size_t pool = (off + 255) & ~(size_t)255;
  _Float16* qh    = (_Float16*)alloc((size_t)BN * NTOK * HD32 * 2);
  _Float16* qcha1 = (_Float16*)alloc((size_t)MTOK * 128 * 2);
  _Float16* qsptb = (_Float16*)alloc((size_t)BN * KSPT * KCH * 2);
  int*      sidxg = (int*)alloc((size_t)BN * KSPT * 4);
  _Float16* vselb = (_Float16*)alloc((size_t)BW * KSPT * CC * 2);
  _Float16* khb   = (_Float16*)alloc((size_t)BN * NTOK * KCH * 2);
  _Float16* vbuf  = (_Float16*)alloc((size_t)BN * KSPT * HD32 * 2);
  _Float16* obuf  = (_Float16*)alloc((size_t)MTOK * CC * 2);
  // overlays: everything from `pool` is dead once the proj GEMM has written out
  _Float16* xn2  = xw;                          // reuses xw (dead after q GEMM)
  _Float16* hbuf = (_Float16*)(ws + pool);      // 205.5 MB inside 229 MB pool

  // weight pre-swizzle into WMMA fragment order (f16)
  swizzleB_kernel<<<(65536 + 255) / 256, 256, 0, stream>>>(wq,  wq_sw,  256, 256);
  swizzleB_kernel<<<(16384 + 255) / 256, 256, 0, stream>>>(wk,  wk_sw,  128, 128);
  swizzleB_kernel<<<(65536 + 255) / 256, 256, 0, stream>>>(wv,  wv_sw,  256, 256);
  swizzleB_kernel<<<(65536 + 255) / 256, 256, 0, stream>>>(wpj, wpj_sw, 256, 256);
  swizzleB_kernel<<<(262144 + 255) / 256, 256, 0, stream>>>(w1, w1_sw, 256, 1024);
  swizzleB_kernel<<<(262144 + 255) / 256, 256, 0, stream>>>(w2, w2_sw, 1024, 256);

  // attention half
  ln_win_kernel<<<MTOK / 8, 256, 0, stream>>>(x, n1g, n1b, xw);
  gemm_k<EpiQ, 2><<<dim3(MTOK / 16, 1), 256, 0, stream>>>(xw, wq_sw, 256,
                                                          EpiQ{bq, qh});
  topk_gather_kernel<<<BW, 256, 0, stream>>>(qh, wch, bch, qcha1, qsptb, vselb, sidxg);
  gemm_k<EpiK, 1><<<dim3(MTOK / 16, 1), 256, 0, stream>>>(qcha1, wk_sw, 128,
                                                          EpiK{bk, khb});
  gemm_k<EpiV, 2><<<dim3(BW * KSPT / 16, 1), 256, 0, stream>>>(vselb, wv_sw, 256,
                                                               EpiV{bv, vbuf});
  attn_kernel<<<BW, 256, 0, stream>>>(khb, qsptb, vbuf, rpb, sidxg, obuf);
  gemm_k<EpiProj, 2><<<dim3(MTOK / 16, 1), 256, 0, stream>>>(obuf, wpj_sw, 256,
                                                             EpiProj{bpj, x, out});
  // MLP half
  ln_flat_kernel<<<MTOK / 8, 256, 0, stream>>>(out, n2g, n2b, xn2);
  gemm_k<EpiM1, 2><<<dim3(MTOK / 16, 4), 256, 0, stream>>>(xn2, w1_sw, 256,
                                                           EpiM1{b1, hbuf});
  gemm_k<EpiM2, 2><<<dim3(MTOK / 16, 1), 256, 0, stream>>>(hbuf, w2_sw, 1024,
                                                           EpiM2{b2, out});
}
